// Head_13503377178782
// MI455X (gfx1250) — matrix operand order repaired and checked by reading, not compiled, not run on hardware
//
#include <hip/hip_runtime.h>
#include <hip/hip_bf16.h>
#include <math.h>

// ---------------------------------------------------------------------------
// Causal single-head attention for MI455X (gfx1250), wave32 + WMMA bf16.
//   x:[4,4096,1024] f32, Wq/Wk/Wv:[1024,64] f32  ->  z:[4,4096,64] f32
// Pipeline: pack_w -> qkv_proj (async-LDS W staging + pipelined WMMA)
//                  -> flash_attn (64-key tiles, online softmax, WMMA)
// ---------------------------------------------------------------------------

#define BATCH 4
#define SEQ   4096
#define DIN   1024
#define HD    64

typedef __attribute__((ext_vector_type(16))) __bf16 v16bf;
typedef __attribute__((ext_vector_type(8)))  __bf16 v8bf;
typedef __attribute__((ext_vector_type(8)))  float  v8f;
typedef int v4i_gcc __attribute__((vector_size(16)));   // matches builtin param

#if defined(__has_builtin)
#if __has_builtin(__builtin_amdgcn_global_load_async_to_lds_b128)
#define ASYNC_STAGING 1
#else
#define ASYNC_STAGING 0
#endif
#else
#define ASYNC_STAGING 0
#endif

__device__ __forceinline__ unsigned short f32_to_bf16_bits(float f) {
  unsigned u = __builtin_bit_cast(unsigned, f);
  unsigned r = (u + 0x7FFFu + ((u >> 16) & 1u)) >> 16;   // RNE
  return (unsigned short)r;
}
__device__ __forceinline__ __bf16 f2bf(float f) {
  return __builtin_bit_cast(__bf16, f32_to_bf16_bits(f));
}
// Load 16 contiguous bf16 (two aligned 16B chunks) into a WMMA fragment reg.
__device__ __forceinline__ v16bf load_bf16x16(const unsigned short* p) {
  v8bf lo = *reinterpret_cast<const v8bf*>(p);
  v8bf hi = *reinterpret_cast<const v8bf*>(p + 8);
  v16bf r;
#pragma unroll
  for (int e = 0; e < 8; ++e) { r[e] = lo[e]; r[e + 8] = hi[e]; }
  return r;
}

// 16-byte global -> LDS copy; async DMA (ASYNCcnt) when available.
__device__ __forceinline__ void copy16_g2l(const unsigned short* g, unsigned short* l) {
#if ASYNC_STAGING
  __builtin_amdgcn_global_load_async_to_lds_b128(
      (__attribute__((address_space(1))) v4i_gcc*)g,
      (__attribute__((address_space(3))) v4i_gcc*)l, 0, 0);
#else
  *reinterpret_cast<uint4*>(l) = *reinterpret_cast<const uint4*>(g);
#endif
}
__device__ __forceinline__ void wait_async_prev() {   // allow 6 (next chunk) in flight
#if ASYNC_STAGING
#if defined(__has_builtin) && __has_builtin(__builtin_amdgcn_s_wait_asynccnt)
  __builtin_amdgcn_s_wait_asynccnt(6);
#else
  asm volatile("s_wait_asynccnt 6" ::: "memory");
#endif
#endif
}
__device__ __forceinline__ void wait_async_all() {
#if ASYNC_STAGING
#if defined(__has_builtin) && __has_builtin(__builtin_amdgcn_s_wait_asynccnt)
  __builtin_amdgcn_s_wait_asynccnt(0);
#else
  asm volatile("s_wait_asynccnt 0" ::: "memory");
#endif
#endif
}

// ---------------------------------------------------------------------------
// Kernel 1: pack W into B-fragment layout, bf16.
// B-fragment (K=32, 16-bit data): lane holds column n = lane%16,
// contraction k = ks*32 + (lane<16 ? 0 : 8) + (e<8 ? e : 8 + e),  e = 0..15.
// frag id = mat*128 + ks*4 + ct  (ct = 16-wide H column tile), 512 u16 each.
// ---------------------------------------------------------------------------
__global__ void pack_w_kernel(const float* __restrict__ Wq,
                              const float* __restrict__ Wk,
                              const float* __restrict__ Wv,
                              unsigned short* __restrict__ wpack) {
  int t    = blockIdx.x * blockDim.x + threadIdx.x;   // 12288 threads
  int lane = t & 31;
  int frag = t >> 5;                                  // 0..383
  int ct   = frag & 3;
  int ks   = (frag >> 2) & 31;
  int mat  = frag >> 7;
  const float* W = (mat == 0) ? Wq : ((mat == 1) ? Wk : Wv);
  int n     = ct * 16 + (lane & 15);
  int kbase = ks * 32 + ((lane < 16) ? 0 : 8);
  alignas(16) unsigned short vals[16];
#pragma unroll
  for (int e = 0; e < 16; ++e)
    vals[e] = f32_to_bf16_bits(W[(size_t)(kbase + (e < 8 ? e : 8 + e)) * HD + n]);
  unsigned short* dst = wpack + ((size_t)frag * 32 + lane) * 16;
  *reinterpret_cast<uint4*>(dst)       = *reinterpret_cast<uint4*>(vals);
  *(reinterpret_cast<uint4*>(dst) + 1) = *(reinterpret_cast<uint4*>(vals) + 1);
}

// ---------------------------------------------------------------------------
// Kernel 2: q/k/v projection.
// 4 waves/block, one 16-row x-tile per wave. Per k-step the 12 shared W
// fragments (12 KB) are staged into LDS once per block — double-buffered via
// GLOBAL_LOAD_ASYNC_TO_LDS_B128 so step ks+1's DMA overlaps step ks's WMMAs.
// x chunk for step ks+1 is global-loaded before step ks's WMMA burst; LDS
// B-fragments run through a 2-deep rotating pipeline so each WMMA overlaps
// the next fragment's ds_load (partial dscnt waits instead of full drains).
// q,k stored row-major bf16 [B,L,64]; v stored transposed bf16 [B,64,L].
// ---------------------------------------------------------------------------
#define WCHUNK 6144   // u16 per staged k-step chunk (12 frags * 512)

__device__ __forceinline__ void stage_w_chunk(const unsigned short* __restrict__ wpack,
                                              unsigned short* lds_dst, int ks, int tid) {
  // 128 threads x 6 x 16B = 12 KB
#pragma unroll
  for (int i = 0; i < 6; ++i) {
    int idx16   = tid * 48 + i * 8;        // u16 index in chunk
    int fl      = idx16 >> 9;              // local frag 0..11
    int within  = idx16 & 511;
    int mat     = fl >> 2;
    int ct      = fl & 3;
    const unsigned short* src =
        wpack + (((size_t)mat * 128 + (size_t)ks * 4 + ct) << 9) + within;
    copy16_g2l(src, lds_dst + idx16);
  }
}

__device__ __forceinline__ v16bf make_a_frag(float4 f0, float4 f1, float4 f2, float4 f3) {
  v16bf a;
  a[0]=f2bf(f0.x); a[1]=f2bf(f0.y); a[2]=f2bf(f0.z); a[3]=f2bf(f0.w);
  a[4]=f2bf(f1.x); a[5]=f2bf(f1.y); a[6]=f2bf(f1.z); a[7]=f2bf(f1.w);
  a[8]=f2bf(f2.x); a[9]=f2bf(f2.y); a[10]=f2bf(f2.z); a[11]=f2bf(f2.w);
  a[12]=f2bf(f3.x); a[13]=f2bf(f3.y); a[14]=f2bf(f3.z); a[15]=f2bf(f3.w);
  return a;
}

__global__ void qkv_proj_kernel(const float* __restrict__ x,
                                const unsigned short* __restrict__ wpack,
                                unsigned short* __restrict__ qbuf,
                                unsigned short* __restrict__ kbuf,
                                unsigned short* __restrict__ vbufT) {
  __shared__ __align__(16) unsigned short wlds[2][WCHUNK];   // 24 KB
  int tid  = threadIdx.x;
  int lane = tid & 31;
  int wave = tid >> 5;
  int tile = blockIdx.x * 4 + wave;          // 0..1023
  int b    = tile >> 8;
  int r0   = (tile & 255) << 4;
  int half = lane >> 4;
  int ln   = lane & 15;

  v8f acc[12];   // [mat*4 + ct]
#pragma unroll
  for (int i = 0; i < 12; ++i) acc[i] = (v8f){0.f,0.f,0.f,0.f,0.f,0.f,0.f,0.f};

  const float* xrow = x + ((size_t)(b * SEQ + r0 + ln)) * DIN + (half ? 8 : 0);

  stage_w_chunk(wpack, &wlds[0][0], 0, tid);

  // preload x chunk for ks = 0
  float4 f0 = *(const float4*)(xrow);
  float4 f1 = *(const float4*)(xrow + 4);
  float4 f2 = *(const float4*)(xrow + 16);
  float4 f3 = *(const float4*)(xrow + 20);

  for (int ks = 0; ks < 32; ++ks) {
    if (ks + 1 < 32) {
      stage_w_chunk(wpack, &wlds[(ks + 1) & 1][0], ks + 1, tid);
      wait_async_prev();    // previous chunk complete; next 6 still in flight
    } else {
      wait_async_all();
    }
    __syncthreads();

    // A fragment for this step from preloaded x registers.
    v16bf a = make_a_frag(f0, f1, f2, f3);

    // Issue next step's x loads now; consumed next iteration (latency hidden
    // behind the 12-WMMA burst below).
    if (ks + 1 < 32) {
      const float* xn = xrow + (ks + 1) * 32;
      f0 = *(const float4*)(xn);
      f1 = *(const float4*)(xn + 4);
      f2 = *(const float4*)(xn + 16);
      f3 = *(const float4*)(xn + 20);
    }

    // 2-deep pipelined LDS fragment loads + WMMA burst.
    const unsigned short* wb = &wlds[ks & 1][0];
    v16bf cur = load_bf16x16(wb + (size_t)lane * 16);
#pragma unroll
    for (int fl = 0; fl < 12; ++fl) {
      v16bf nxt;
      if (fl < 11) nxt = load_bf16x16(wb + ((size_t)(fl + 1) * 32 + lane) * 16);
      acc[fl] = __builtin_amdgcn_wmma_f32_16x16x32_bf16(
          false, a, false, cur, (short)0, acc[fl], false, false);
      if (fl < 11) cur = nxt;
    }

    __syncthreads();   // all waves done reading before this buffer is restaged
  }

  // C/D layout: VGPR r, lane -> row = r + 8*half, col = lane%16 (+16*ct).
#pragma unroll
  for (int ct = 0; ct < 4; ++ct) {
    int h = ct * 16 + ln;
#pragma unroll
    for (int r = 0; r < 8; ++r) {
      int row = r0 + r + half * 8;
      size_t idx = ((size_t)(b * SEQ + row)) * HD + h;
      qbuf[idx] = f32_to_bf16_bits(acc[0 * 4 + ct][r]);
      kbuf[idx] = f32_to_bf16_bits(acc[1 * 4 + ct][r]);
      vbufT[((size_t)b * HD + h) * SEQ + row] = f32_to_bf16_bits(acc[2 * 4 + ct][r]);
    }
  }
}

// ---------------------------------------------------------------------------
// Kernel 3: flash attention, 64-key tiles. One wave per 16-row query tile.
// S = Q K^T : 8 WMMAs (batched K fragments). Online softmax via width-16
// half-wave shuffle reductions. P re-shaped C-layout -> A-layout through a
// per-wave 2KB LDS buffer, O += P V : 8 WMMAs (batched V fragments).
// ---------------------------------------------------------------------------
__global__ void flash_attn_kernel(const unsigned short* __restrict__ qbuf,
                                  const unsigned short* __restrict__ kbuf,
                                  const unsigned short* __restrict__ vbufT,
                                  float* __restrict__ out) {
  __shared__ __align__(16) unsigned short pbuf[4][16][64];   // 8 KB
  int lane = threadIdx.x & 31;
  int wave = threadIdx.x >> 5;
  int tile = blockIdx.x * 4 + wave;          // 0..1023
  int b    = tile >> 8;
  int q0   = (tile & 255) << 4;
  int half = lane >> 4;
  int ln   = lane & 15;
  size_t bL = (size_t)b * SEQ;

  // Q A-fragments for the two 32-wide h chunks.
  const unsigned short* qrow = qbuf + (bL + q0 + ln) * HD;
  v16bf aq[2];
#pragma unroll
  for (int s = 0; s < 2; ++s) {
    const unsigned short* p0 = qrow + 32 * s + (half ? 16 : 0);
    v8bf lo = *reinterpret_cast<const v8bf*>(p0);
    v8bf hi = *reinterpret_cast<const v8bf*>(p0 + 8);
#pragma unroll
    for (int e = 0; e < 8; ++e) { aq[s][e] = lo[e]; aq[s][e + 8] = hi[e]; }
  }

  v8f o[4];
#pragma unroll
  for (int c = 0; c < 4; ++c) o[c] = (v8f){0.f,0.f,0.f,0.f,0.f,0.f,0.f,0.f};
  float m_i[8], l_i[8];
#pragma unroll
  for (int r = 0; r < 8; ++r) { m_i[r] = -1e30f; l_i[r] = 0.f; }

  const int ktmax = (q0 + 15) >> 6;                 // diagonal tile == last tile
  const float kscale = 0.125f * 1.44269504088896f;  // 1/sqrt(64) * log2(e)

  for (int kt = 0; kt <= ktmax; ++kt) {
    size_t k0 = bL + (size_t)kt * 64;

    // ---- batched K fragments, then S = Q K^T (8 WMMAs) ----
    v16bf bk[8];
#pragma unroll
    for (int nt = 0; nt < 4; ++nt)
#pragma unroll
      for (int s = 0; s < 2; ++s)
        bk[nt * 2 + s] =
            load_bf16x16(kbuf + (k0 + nt * 16 + ln) * HD + 32 * s + half * 16);
    v8f sc[4];
#pragma unroll
    for (int nt = 0; nt < 4; ++nt) {
      sc[nt] = (v8f){0.f,0.f,0.f,0.f,0.f,0.f,0.f,0.f};
#pragma unroll
      for (int s = 0; s < 2; ++s)
        sc[nt] = __builtin_amdgcn_wmma_f32_16x16x32_bf16(
            false, aq[s], false, bk[nt * 2 + s], (short)0, sc[nt], false, false);
    }
#pragma unroll
    for (int nt = 0; nt < 4; ++nt)
#pragma unroll
      for (int r = 0; r < 8; ++r) sc[nt][r] *= kscale;

    if (kt == ktmax) {                // causal mask, only the diagonal tile
#pragma unroll
      for (int nt = 0; nt < 4; ++nt) {
        int kp = kt * 64 + nt * 16 + ln;
#pragma unroll
        for (int r = 0; r < 8; ++r)
          if (kp > q0 + r + half * 8) sc[nt][r] = -1e30f;
      }
    }

    // ---- online softmax (width-16 half-wave reductions) ----
    float alpha[8];
#pragma unroll
    for (int r = 0; r < 8; ++r) {
      float mx = fmaxf(fmaxf(sc[0][r], sc[1][r]), fmaxf(sc[2][r], sc[3][r]));
      mx = fmaxf(mx, __shfl_xor(mx, 1, 16));
      mx = fmaxf(mx, __shfl_xor(mx, 2, 16));
      mx = fmaxf(mx, __shfl_xor(mx, 4, 16));
      mx = fmaxf(mx, __shfl_xor(mx, 8, 16));
      float mn = fmaxf(m_i[r], mx);
      float a_ = exp2f(m_i[r] - mn);
      int row = r + half * 8;
      float rs = 0.f;
#pragma unroll
      for (int nt = 0; nt < 4; ++nt) {
        float p = exp2f(sc[nt][r] - mn);
        rs += p;
        pbuf[wave][row][nt * 16 + ln] = f32_to_bf16_bits(p);
      }
      rs += __shfl_xor(rs, 1, 16);
      rs += __shfl_xor(rs, 2, 16);
      rs += __shfl_xor(rs, 4, 16);
      rs += __shfl_xor(rs, 8, 16);
      m_i[r] = mn;
      l_i[r] = l_i[r] * a_ + rs;
      alpha[r] = a_;
    }
#pragma unroll
    for (int c = 0; c < 4; ++c)
#pragma unroll
      for (int r = 0; r < 8; ++r) o[c][r] *= alpha[r];

    // per-wave LDS staging: stores landed before the transposed reads
    asm volatile("s_wait_dscnt 0" ::: "memory");

    // ---- P as two A-fragments (16x32 each, keypos contraction) ----
    v16bf pa[2];
#pragma unroll
    for (int c = 0; c < 2; ++c) {
      const unsigned short* pl0 = &pbuf[wave][ln][32 * c + (half ? 16 : 0)];
      v8bf lo = *reinterpret_cast<const v8bf*>(pl0);
      v8bf hi = *reinterpret_cast<const v8bf*>(pl0 + 8);
#pragma unroll
      for (int e = 0; e < 8; ++e) { pa[c][e] = lo[e]; pa[c][e + 8] = hi[e]; }
    }

    // ---- batched V fragments (transposed: contiguous keypos), 8 WMMAs ----
    v16bf bv[8];
#pragma unroll
    for (int ct = 0; ct < 4; ++ct)
#pragma unroll
      for (int c = 0; c < 2; ++c)
        bv[ct * 2 + c] = load_bf16x16(
            vbufT + ((size_t)b * HD + ct * 16 + ln) * SEQ +
            (size_t)kt * 64 + 32 * c + half * 16);
#pragma unroll
    for (int ct = 0; ct < 4; ++ct)
#pragma unroll
      for (int c = 0; c < 2; ++c)
        o[ct] = __builtin_amdgcn_wmma_f32_16x16x32_bf16(
            false, pa[c], false, bv[ct * 2 + c], (short)0, o[ct], false, false);

    if (kt < ktmax) {  // prefetch next K tile (global_prefetch_b8)
      __builtin_prefetch(kbuf + (bL + (size_t)(kt + 1) * 64 + ln) * HD, 0, 1);
      __builtin_prefetch(vbufT + ((size_t)b * HD + ln) * SEQ + (size_t)(kt + 1) * 64, 0, 1);
    }
  }

  // ---- epilogue: O /= l, store f32 ----
  float inv[8];
#pragma unroll
  for (int r = 0; r < 8; ++r) inv[r] = 1.0f / l_i[r];
#pragma unroll
  for (int ct = 0; ct < 4; ++ct) {
    int h = ct * 16 + ln;
#pragma unroll
    for (int r = 0; r < 8; ++r) {
      int row = q0 + r + half * 8;
      out[(bL + row) * HD + h] = o[ct][r] * inv[r];
    }
  }
}

// ---------------------------------------------------------------------------
extern "C" void kernel_launch(void* const* d_in, const int* in_sizes, int n_in,
                              void* d_out, int out_size, void* d_ws, size_t ws_size,
                              hipStream_t stream) {
  const float* x  = (const float*)d_in[0];
  const float* Wq = (const float*)d_in[1];
  const float* Wk = (const float*)d_in[2];
  const float* Wv = (const float*)d_in[3];
  float* z = (float*)d_out;

  // workspace layout (bf16 as u16), all 256B-aligned:
  //   wpack : 3*32*4*32*16 u16 = 393216 B
  //   qbuf  : 4*4096*64 u16 = 2 MB
  //   kbuf  : 2 MB
  //   vbufT : 2 MB (transposed [B,64,L])
  unsigned short* wpack = (unsigned short*)d_ws;
  unsigned short* qbuf  = (unsigned short*)((char*)d_ws + 393216);
  unsigned short* kbuf  = qbuf + (size_t)BATCH * SEQ * HD;
  unsigned short* vbufT = kbuf + (size_t)BATCH * SEQ * HD;

  pack_w_kernel<<<96, 128, 0, stream>>>(Wq, Wk, Wv, wpack);
  qkv_proj_kernel<<<256, 128, 0, stream>>>(x, wpack, qbuf, kbuf, vbufT);
  flash_attn_kernel<<<256, 128, 0, stream>>>(qbuf, kbuf, vbufT, z);
}
